// static_heto_graph_23192823399235
// MI455X (gfx1250) — matrix operand order repaired and checked by reading, not compiled, not run on hardware
//
#include <hip/hip_runtime.h>
#include <hip/hip_bf16.h>
#include <math.h>

typedef __attribute__((ext_vector_type(16))) _Float16 v16h;
typedef __attribute__((ext_vector_type(8)))  _Float16 v8h;
typedef __attribute__((ext_vector_type(8)))  float    v8f;

#define N_WORD  240000
#define N_TOPIC 800
#define N_DOC   1600
#define H_INP   100
#define KPAD    128
#define H_DIM   64
#define NB      16
#define E_WW 1600000
#define E_WT 800000
#define E_WD 800000
#define E_TD 160000
#define E_TT 80000

// ---------------------------------------------------------------------------
// Gather word embeddings -> f16, padded to KPAD columns (zeros beyond H_INP)
// ---------------------------------------------------------------------------
__global__ void gather_word_h(const float* __restrict__ emb, const int* __restrict__ ids,
                              _Float16* __restrict__ out) {
  long i = (long)blockIdx.x * blockDim.x + threadIdx.x;
  if (i >= (long)N_WORD * KPAD) return;
  long n = i >> 7;                 // /128
  int  d = (int)(i & (KPAD - 1));
  out[i] = (d < H_INP) ? (_Float16)emb[(long)ids[n] * H_INP + d] : (_Float16)0.0f;
}

__global__ void gather_topic_h(const float* __restrict__ emb, const int* __restrict__ ids,
                               _Float16* __restrict__ out) {
  int i = blockIdx.x * blockDim.x + threadIdx.x;
  if (i >= N_TOPIC * H_DIM) return;
  int n = i >> 6, d = i & 63;
  out[i] = (_Float16)emb[ids[n] * H_DIM + d];
}

// ---------------------------------------------------------------------------
// Weight prep: W[Kin x 64] f32 -> Wt[64 x Kpad] f16 (transposed, zero-padded)
// ---------------------------------------------------------------------------
template <int KP>
__global__ void prep_weight_t(const float* __restrict__ W, _Float16* __restrict__ Wt, int Kin) {
  int i = blockIdx.x * blockDim.x + threadIdx.x;
  if (i >= 64 * KP) return;
  int n = i / KP, k = i - n * KP;
  Wt[i] = (k < Kin) ? (_Float16)W[k * 64 + n] : (_Float16)0.0f;
}

// ---------------------------------------------------------------------------
// WMMA GEMM: Y[M x 64] (f32) = X[M x K] (f16) @ Wt^T (Wt is [64 x K] f16) + bias
// block = 256 = 8 waves = 2 M-tiles x 4 N-tiles. M multiple of 32, K in {64,128}.
// A frag (16b 16x32): lane<16 holds K={k0..k0+7, k0+16..k0+23} of row lane&15.
// B frag (16b 32x16): lane<16 holds K={k0..k0+7, k0+16..k0+23} of col lane&15.
// C/D frag: VGPR v -> row v + 8*(lane>=16), col lane&15.
// ---------------------------------------------------------------------------
template <int K>
__global__ void gemm_wmma_h(const _Float16* __restrict__ X, const _Float16* __restrict__ Wt,
                            const float* __restrict__ bias, float* __restrict__ Y) {
  const int lane  = threadIdx.x & 31;
  const int wave  = threadIdx.x >> 5;
  const int mTile = blockIdx.x * 2 + (wave >> 2);
  const int nTile = wave & 3;
  const int r15   = lane & 15;
  const int hi    = lane >> 4;            // 0 lower half, 1 upper half
  const int row   = mTile * 16 + r15;
  const int col   = nTile * 16 + r15;

  const _Float16* __restrict__ xr = X + (long)row * K + hi * 8;
  const _Float16* __restrict__ wr = Wt + (long)col * K + hi * 8;

  v8f acc = {};
#pragma unroll
  for (int k0 = 0; k0 < K; k0 += 32) {
    v8h a0 = *(const v8h*)(xr + k0);        // K = k0+half8 .. +8
    v8h a1 = *(const v8h*)(xr + k0 + 16);   // K = k0+16+half8 .. +8
    v8h b0 = *(const v8h*)(wr + k0);        // K = k0+half8 .. +8
    v8h b1 = *(const v8h*)(wr + k0 + 16);   // K = k0+16+half8 .. +8
    v16h a = __builtin_shufflevector(a0, a1, 0,1,2,3,4,5,6,7,8,9,10,11,12,13,14,15);
    v16h b = __builtin_shufflevector(b0, b1, 0,1,2,3,4,5,6,7,8,9,10,11,12,13,14,15);
    acc = __builtin_amdgcn_wmma_f32_16x16x32_f16(false, a, false, b,
                                                 (short)0, acc, false, false);
  }

  const float bv    = bias[col];
  const int   mBase = mTile * 16 + hi * 8;
#pragma unroll
  for (int v = 0; v < 8; ++v) {
    Y[(long)(mBase + v) * 64 + col] = acc[v] + bv;
  }
}

// ---------------------------------------------------------------------------
// Edge scatter: one wave per edge, 2 dims per lane (f32 atomics).
// ---------------------------------------------------------------------------
__global__ void scatter_kernel(const float* __restrict__ msg, const int* __restrict__ src,
                               const int* __restrict__ dst, const float* __restrict__ w,
                               int E, float* __restrict__ acc, float* __restrict__ cnt) {
  long t = (long)blockIdx.x * blockDim.x + threadIdx.x;
  int  e = (int)(t >> 5);
  if (e >= E) return;
  int   lane = threadIdx.x & 31;
  int   s    = src[e];
  int   d    = dst[e];
  float we   = w[e];
  float m0 = msg[(long)s * 64 + lane];
  float m1 = msg[(long)s * 64 + lane + 32];
  atomicAdd(&acc[(long)d * 64 + lane],      m0 * we);
  atomicAdd(&acc[(long)d * 64 + lane + 32], m1 * we);
  if (lane == 0) atomicAdd(&cnt[d], 1.0f);
}

// ---------------------------------------------------------------------------
__global__ void zero_kernel(float* __restrict__ p, long n) {
  long i = (long)blockIdx.x * blockDim.x + threadIdx.x;
  if (i < n) p[i] = 0.0f;
}

// mean -> f16 (for GEMM consumption)
__global__ void mean_to_half(const float* __restrict__ acc, const float* __restrict__ cnt,
                             _Float16* __restrict__ out, long n64) {
  long i = (long)blockIdx.x * blockDim.x + threadIdx.x;
  if (i >= n64) return;
  out[i] = (_Float16)(acc[i] / fmaxf(cnt[i >> 6], 1.0f));
}

__global__ void leaky_half(_Float16* __restrict__ p, long n) {
  long i = (long)blockIdx.x * blockDim.x + threadIdx.x;
  if (i >= n) return;
  float v = (float)p[i];
  p[i] = (_Float16)(v > 0.0f ? v : 0.01f * v);
}

// two seg-means summed, leaky, -> f16 (t1)
__global__ void combine2_to_half(const float* __restrict__ a1, const float* __restrict__ c1,
                                 const float* __restrict__ a2, const float* __restrict__ c2,
                                 _Float16* __restrict__ out, long n64) {
  long i = (long)blockIdx.x * blockDim.x + threadIdx.x;
  if (i >= n64) return;
  float v = a1[i] / fmaxf(c1[i >> 6], 1.0f) + a2[i] / fmaxf(c2[i >> 6], 1.0f);
  out[i] = (_Float16)(v > 0.0f ? v : 0.01f * v);
}

// two seg-means summed -> f32 (d2)
__global__ void combine2_f32(const float* __restrict__ a1, const float* __restrict__ c1,
                             const float* __restrict__ a2, const float* __restrict__ c2,
                             float* __restrict__ out, long n64) {
  long i = (long)blockIdx.x * blockDim.x + threadIdx.x;
  if (i >= n64) return;
  out[i] = a1[i] / fmaxf(c1[i >> 6], 1.0f) + a2[i] / fmaxf(c2[i >> 6], 1.0f);
}

// ---------------------------------------------------------------------------
// Final: doc pooling (100 docs/graph), logits, BCE loss, sigmoid.
// ---------------------------------------------------------------------------
__global__ void final_kernel(const float* __restrict__ d2, const float* __restrict__ ow,
                             const float* __restrict__ ob, const float* __restrict__ y,
                             float* __restrict__ out) {
  __shared__ float logits[NB];
  int b    = threadIdx.x >> 5;
  int lane = threadIdx.x & 31;
  float w0 = ow[lane], w1 = ow[lane + 32];
  float p = 0.0f;
  for (int doc = 0; doc < 100; ++doc) {
    const float* row = d2 + (long)(b * 100 + doc) * 64;
    p += row[lane] * w0 + row[lane + 32] * w1;
  }
#pragma unroll
  for (int off = 16; off > 0; off >>= 1) p += __shfl_down(p, off, 32);
  if (lane == 0) logits[b] = p * 0.01f + ob[0];
  __syncthreads();
  if (threadIdx.x == 0) {
    float loss = 0.0f;
    for (int i = 0; i < NB; ++i) {
      float x = logits[i];
      loss += fmaxf(x, 0.0f) + log1pf(expf(-fabsf(x))) - x * y[i];
      out[1 + i] = 1.0f / (1.0f + expf(-x));
    }
    out[0] = loss / (float)NB;
  }
}

// ---------------------------------------------------------------------------
static inline unsigned cdiv(long a, long b) { return (unsigned)((a + b - 1) / b); }

extern "C" void kernel_launch(void* const* d_in, const int* in_sizes, int n_in,
                              void* d_out, int out_size, void* d_ws, size_t ws_size,
                              hipStream_t stream) {
  // ---- inputs (setup_inputs dict order) ----
  const int*   word_id  = (const int*)d_in[0];
  const int*   topic_id = (const int*)d_in[1];
  const int*   src_ww = (const int*)d_in[2];
  const int*   dst_ww = (const int*)d_in[3];
  const float* w_ww   = (const float*)d_in[4];
  const int*   src_wt = (const int*)d_in[5];
  const int*   dst_wt = (const int*)d_in[6];
  const float* w_wt   = (const float*)d_in[7];
  const int*   src_wd = (const int*)d_in[8];
  const int*   dst_wd = (const int*)d_in[9];
  const float* w_wd   = (const float*)d_in[10];
  const int*   src_td = (const int*)d_in[11];
  const int*   dst_td = (const int*)d_in[12];
  const float* w_td   = (const float*)d_in[13];
  const int*   src_tt = (const int*)d_in[14];
  const int*   dst_tt = (const int*)d_in[15];
  const float* w_tt   = (const float*)d_in[16];
  const float* y_data = (const float*)d_in[17];
  // params, jax pytree flatten (sorted keys): l1{td,tt,wd,wt,ww}{b,w}, l2{...},
  // out{b,w}, topic_embeds, word_embeds
  const float* l1_td_b = (const float*)d_in[18];  (void)l1_td_b;
  const float* l1_td_w = (const float*)d_in[19];  (void)l1_td_w;
  const float* l1_tt_b = (const float*)d_in[20];
  const float* l1_tt_w = (const float*)d_in[21];
  const float* l1_wt_b = (const float*)d_in[24];
  const float* l1_wt_w = (const float*)d_in[25];
  const float* l1_ww_b = (const float*)d_in[26];
  const float* l1_ww_w = (const float*)d_in[27];
  const float* l2_td_b = (const float*)d_in[28];
  const float* l2_td_w = (const float*)d_in[29];
  const float* l2_wd_b = (const float*)d_in[32];
  const float* l2_wd_w = (const float*)d_in[33];
  const float* l2_ww_b = (const float*)d_in[36];
  const float* l2_ww_w = (const float*)d_in[37];
  const float* out_b   = (const float*)d_in[38];
  const float* out_w   = (const float*)d_in[39];
  const float* topic_embeds = (const float*)d_in[40];
  const float* word_embeds  = (const float*)d_in[41];
  (void)in_sizes; (void)n_in; (void)out_size; (void)ws_size;

  // ---- workspace layout (float-sized units; all blocks 16B aligned) ----
  float* wsf = (float*)d_ws;
  // region0: h_wordf [240000x128 f16] == 15,360,000 f32 slots; reused as f32 acc (layer2 ww)
  _Float16* h_wordf = (_Float16*)wsf;
  float*    acc2    = wsf;                              // alias (used after h_wordf dead)
  float*    msg     = wsf + 15360000L;                  // 15,360,000 f32
  float*    acc1    = msg + 15360000L;                  // 15,360,000 f32
  float*    cntw    = acc1 + 15360000L;                 // 240,000 f32
  _Float16* h_wh    = (_Float16*)(cntw + 240000L);      // 240000x64 f16 = 7,680,000 f32 slots
  float*    p0      = cntw + 240000L + 7680000L;
  _Float16* h_topicf= (_Float16*)p0;                    // 51,200 f16 = 25,600 f32
  _Float16* h_t1    = (_Float16*)(p0 + 25600L);         // 25,600 f32
  float*    acc_t1  = p0 + 51200L;                      // 51,200
  float*    acc_t2  = acc_t1 + 51200L;                  // 51,200
  float*    cnt_t1  = acc_t2 + 51200L;                  // 800
  float*    cnt_t2  = cnt_t1 + 800L;                    // 800
  float*    acc_d1  = cnt_t2 + 800L;                    // 102,400
  float*    acc_d2  = acc_d1 + 102400L;                 // 102,400
  float*    cnt_d1  = acc_d2 + 102400L;                 // 1,600
  float*    cnt_d2  = cnt_d1 + 1600L;                   // 1,600
  float*    d2b     = cnt_d2 + 1600L;                   // 102,400
  float*    pw      = d2b + 102400L;
  _Float16* wt_l1ww = (_Float16*)pw;                    // 64*128 f16 = 4096 f32
  _Float16* wt_l1wt = (_Float16*)(pw + 4096L);          // 64*64 f16 = 2048 f32
  _Float16* wt_l1tt = (_Float16*)(pw + 6144L);
  _Float16* wt_l2ww = (_Float16*)(pw + 8192L);
  _Float16* wt_l2wd = (_Float16*)(pw + 10240L);
  _Float16* wt_l2td = (_Float16*)(pw + 12288L);

  const long NW64 = (long)N_WORD * 64;
  const long NT64 = (long)N_TOPIC * 64;
  const long ND64 = (long)N_DOC * 64;

  // ---- weight prep (f32 -> f16, transposed, K-padded) ----
  prep_weight_t<KPAD><<<cdiv(64 * KPAD, 256), 256, 0, stream>>>(l1_ww_w, wt_l1ww, H_INP);
  prep_weight_t<H_DIM><<<cdiv(64 * H_DIM, 256), 256, 0, stream>>>(l1_wt_w, wt_l1wt, H_DIM);
  prep_weight_t<H_DIM><<<cdiv(64 * H_DIM, 256), 256, 0, stream>>>(l1_tt_w, wt_l1tt, H_DIM);
  prep_weight_t<H_DIM><<<cdiv(64 * H_DIM, 256), 256, 0, stream>>>(l2_ww_w, wt_l2ww, H_DIM);
  prep_weight_t<H_DIM><<<cdiv(64 * H_DIM, 256), 256, 0, stream>>>(l2_wd_w, wt_l2wd, H_DIM);
  prep_weight_t<H_DIM><<<cdiv(64 * H_DIM, 256), 256, 0, stream>>>(l2_td_w, wt_l2td, H_DIM);

  // ================= Layer 1 =================
  gather_word_h<<<cdiv((long)N_WORD * KPAD, 256), 256, 0, stream>>>(word_embeds, word_id, h_wordf);
  gather_topic_h<<<cdiv(NT64, 256), 256, 0, stream>>>(topic_embeds, topic_id, h_topicf);

  // msg = lin(l1.ww, word_f)
  gemm_wmma_h<KPAD><<<N_WORD / 32, 256, 0, stream>>>(h_wordf, wt_l1ww, l1_ww_b, msg);
  // word_h = seg_mean(msg[src]*w, dst_ww)  -> f16 h_wh
  zero_kernel<<<cdiv(NW64, 256), 256, 0, stream>>>(acc1, NW64);
  zero_kernel<<<cdiv(N_WORD, 256), 256, 0, stream>>>(cntw, N_WORD);
  scatter_kernel<<<cdiv((long)E_WW * 32, 256), 256, 0, stream>>>(msg, src_ww, dst_ww, w_ww, E_WW, acc1, cntw);
  mean_to_half<<<cdiv(NW64, 256), 256, 0, stream>>>(acc1, cntw, h_wh, NW64);

  // topic_h = seg_mean(lin(l1.wt, word_h)[wt]) + seg_mean(lin(l1.tt, topic_f)[tt]); leaky -> t1
  gemm_wmma_h<H_DIM><<<N_WORD / 32, 256, 0, stream>>>(h_wh, wt_l1wt, l1_wt_b, msg);
  zero_kernel<<<cdiv(NT64, 256), 256, 0, stream>>>(acc_t1, NT64);
  zero_kernel<<<cdiv(N_TOPIC, 256), 256, 0, stream>>>(cnt_t1, N_TOPIC);
  scatter_kernel<<<cdiv((long)E_WT * 32, 256), 256, 0, stream>>>(msg, src_wt, dst_wt, w_wt, E_WT, acc_t1, cnt_t1);

  gemm_wmma_h<H_DIM><<<N_TOPIC / 32, 256, 0, stream>>>(h_topicf, wt_l1tt, l1_tt_b, msg);
  zero_kernel<<<cdiv(NT64, 256), 256, 0, stream>>>(acc_t2, NT64);
  zero_kernel<<<cdiv(N_TOPIC, 256), 256, 0, stream>>>(cnt_t2, N_TOPIC);
  scatter_kernel<<<cdiv((long)E_TT * 32, 256), 256, 0, stream>>>(msg, src_tt, dst_tt, w_tt, E_TT, acc_t2, cnt_t2);

  combine2_to_half<<<cdiv(NT64, 256), 256, 0, stream>>>(acc_t1, cnt_t1, acc_t2, cnt_t2, h_t1, NT64);

  // w1 = leaky(word_h) in place  (layer-1 wd/td feed only unused d1 -> skipped)
  leaky_half<<<cdiv(NW64, 256), 256, 0, stream>>>(h_wh, NW64);

  // ================= Layer 2 =================
  // word_h2 = seg_mean(lin(l2.ww, w1)[ww])  (acc reuses the h_wordf region; h_wordf is dead)
  gemm_wmma_h<H_DIM><<<N_WORD / 32, 256, 0, stream>>>(h_wh, wt_l2ww, l2_ww_b, msg);
  zero_kernel<<<cdiv(NW64, 256), 256, 0, stream>>>(acc2, NW64);
  zero_kernel<<<cdiv(N_WORD, 256), 256, 0, stream>>>(cntw, N_WORD);
  scatter_kernel<<<cdiv((long)E_WW * 32, 256), 256, 0, stream>>>(msg, src_ww, dst_ww, w_ww, E_WW, acc2, cntw);
  mean_to_half<<<cdiv(NW64, 256), 256, 0, stream>>>(acc2, cntw, h_wh, NW64);   // h_wh := word_h2

  // d2 = seg_mean(lin(l2.wd, word_h2)[wd]) + seg_mean(lin(l2.td, t1)[td])
  gemm_wmma_h<H_DIM><<<N_WORD / 32, 256, 0, stream>>>(h_wh, wt_l2wd, l2_wd_b, msg);
  zero_kernel<<<cdiv(ND64, 256), 256, 0, stream>>>(acc_d1, ND64);
  zero_kernel<<<cdiv(N_DOC, 256), 256, 0, stream>>>(cnt_d1, N_DOC);
  scatter_kernel<<<cdiv((long)E_WD * 32, 256), 256, 0, stream>>>(msg, src_wd, dst_wd, w_wd, E_WD, acc_d1, cnt_d1);

  gemm_wmma_h<H_DIM><<<N_TOPIC / 32, 256, 0, stream>>>(h_t1, wt_l2td, l2_td_b, msg);
  zero_kernel<<<cdiv(ND64, 256), 256, 0, stream>>>(acc_d2, ND64);
  zero_kernel<<<cdiv(N_DOC, 256), 256, 0, stream>>>(cnt_d2, N_DOC);
  scatter_kernel<<<cdiv((long)E_TD * 32, 256), 256, 0, stream>>>(msg, src_td, dst_td, w_td, E_TD, acc_d2, cnt_d2);

  combine2_f32<<<cdiv(ND64, 256), 256, 0, stream>>>(acc_d1, cnt_d1, acc_d2, cnt_d2, d2b, ND64);

  // ================= Readout =================
  final_kernel<<<1, NB * 32, 0, stream>>>(d2b, out_w, out_b, y_data, (float*)d_out);
}
